// RNN_31928786878902
// MI455X (gfx1250) — matrix operand order repaired and checked by reading, not compiled, not run on hardware
//
#include <hip/hip_runtime.h>
#include <hip/hip_bf16.h>
#include <stdint.h>

#define N_CELL   4096
#define T_EPOCH  500
#define T_TOTAL  2500
#define NBLOCKS  256           // one wave per block -> one output tile per block
#define NTHREADS 32

typedef __bf16        bf16x16 __attribute__((ext_vector_type(16)));
typedef float         f32x8   __attribute__((ext_vector_type(8)));
typedef float         f32x4   __attribute__((ext_vector_type(4)));
typedef unsigned int  u32x4   __attribute__((ext_vector_type(4)));

union V16BF { bf16x16 v; u32x4 q[2]; };

__device__ __forceinline__ unsigned short f2bf(float f) {
  unsigned u = __float_as_uint(f);
  u = (u + 0x7FFFu + ((u >> 16) & 1u)) >> 16;   // round-to-nearest-even
  return (unsigned short)u;
}

// ---------------------------------------------------------------- init
__global__ void rnn_init(unsigned* __restrict__ counter,
                         float*    __restrict__ accum,
                         unsigned* __restrict__ abuf_u32) {
  const int tid = threadIdx.x;
  if (tid == 0) *counter = 0u;
  if (tid < 16) accum[tid] = 0.0f;
  // zero the 4x4096 bf16 state buffer
  const int nwords = (4 * N_CELL * 2) / 4;      // 8192 u32
  for (int i = tid; i < nwords; i += blockDim.x) abuf_u32[i] = 0u;
}

// ------------------------------------------------- input drives x @ W_in^T
__global__ void rnn_drive(const float* __restrict__ x,      // [4,2,2]
                          const float* __restrict__ inp_w,  // [N_CELL,2]
                          float* __restrict__ ds,           // [4,N_CELL]
                          float* __restrict__ dt) {         // [4,N_CELL]
  const int n = blockIdx.x * blockDim.x + threadIdx.x;
  if (n >= N_CELL) return;
  const float w0 = inp_w[n * 2 + 0];
  const float w1 = inp_w[n * 2 + 1];
#pragma unroll
  for (int m = 0; m < 4; ++m) {
    ds[m * N_CELL + n] = x[m * 4 + 0] * w0 + x[m * 4 + 1] * w1;  // x[m,0,:]
    dt[m * N_CELL + n] = x[m * 4 + 2] * w0 + x[m * 4 + 3] * w1;  // x[m,1,:]
  }
}

// ------------------------------------------------- fp32 -> bf16 weights
// fp32 source is single-use: non-temporal loads keep it from evicting the
// L2-resident bf16 working set we are about to build.
__global__ void rnn_cvt(const float* __restrict__ w,
                        unsigned short* __restrict__ wb) {
  const int i = (blockIdx.x * blockDim.x + threadIdx.x) * 8;
  f32x4 a = __builtin_nontemporal_load((const f32x4*)(w + i));
  f32x4 b = __builtin_nontemporal_load((const f32x4*)(w + i + 4));
  u32x4 o;
  o.x = (unsigned)f2bf(a.x) | ((unsigned)f2bf(a.y) << 16);
  o.y = (unsigned)f2bf(a.z) | ((unsigned)f2bf(a.w) << 16);
  o.z = (unsigned)f2bf(b.x) | ((unsigned)f2bf(b.y) << 16);
  o.w = (unsigned)f2bf(b.z) | ((unsigned)f2bf(b.w) << 16);
  *(u32x4*)(wb + i) = o;
}

// ------------------------------------------------- persistent RNN kernel
__global__ void __launch_bounds__(NTHREADS, 1)
rnn_persist(const unsigned short* __restrict__ Wb,   // [N,N] bf16, row-major (out,in)
            unsigned short*       __restrict__ Ab,   // [4,N] bf16 state (batch rows)
            const float* __restrict__ rec_b,
            const float* __restrict__ ds,
            const float* __restrict__ dt,
            const float* __restrict__ mem_w,         // [2,N]
            const float* __restrict__ out_w,         // [2,N]
            const float* __restrict__ noise,         // [T_TOTAL,N]
            unsigned* __restrict__ counter,
            float*    __restrict__ accum) {
  const int lane = threadIdx.x & 31;
  const int tile = blockIdx.x;                 // 0..255, one 16-neuron tile per wave
  const int nn   = (tile << 4) + (lane & 15);  // this lane's neuron
  const int hi   = (lane >= 16);

  // A-operand row for this lane is M = lane&15, but D rows 4..15 are never read,
  // so rows 4..15 of A may hold anything: feed them duplicates of the 4 live batch
  // rows (lane&3). Identical addresses across lanes coalesce -> ~4x less A traffic,
  // and no divergent predication in the hot loop.
  const int arow = lane & 3;

  // per-lane constants (only lanes 0..15 carry live outputs)
  float bias = 0.f, mw0 = 0.f, mw1 = 0.f, ow0 = 0.f, ow1 = 0.f;
  float dsv[4] = {0.f, 0.f, 0.f, 0.f};
  float dtv[4] = {0.f, 0.f, 0.f, 0.f};
  if (lane < 16) {
    bias = rec_b[nn];
#pragma unroll
    for (int m = 0; m < 4; ++m) {
      dsv[m] = ds[m * N_CELL + nn];
      dtv[m] = dt[m * N_CELL + nn];
    }
    mw0 = mem_w[nn]; mw1 = mem_w[N_CELL + nn];
    ow0 = out_w[nn]; ow1 = out_w[N_CELL + nn];
  }

  float rr[4]  = {0.f, 0.f, 0.f, 0.f};                       // recurrent state (fp32)
  float rmv[8] = {0.f,0.f,0.f,0.f,0.f,0.f,0.f,0.f};          // delay readout sums
  float rsv[8] = {0.f,0.f,0.f,0.f,0.f,0.f,0.f,0.f};          // response readout sums

  // ISA-layout-matched base pointers (row-major buffers, no repack needed):
  //  A 16-bit 16x32: lane l<16 -> row M=l, K = k0+0..7 & k0+16..23; lane l+16 -> K+8
  //  B 16-bit 32x16: lane l -> col l&15, K = k0+0..7 & k0+16..23; lane l+16 -> K+8
  const unsigned short* Abase = Ab + arow * N_CELL + (hi ? 8 : 0);
  const unsigned short* Bbase = Wb + (size_t)nn * N_CELL + (hi ? 8 : 0);

  for (int t = 0; t < T_TOTAL; ++t) {
    f32x8 acc = {0.f, 0.f, 0.f, 0.f, 0.f, 0.f, 0.f, 0.f};
    // One wave per SIMD: latency hiding is purely in-wave ILP, so unroll deep
    // to keep many b128 loads in flight between partial s_wait_loadcnt points.
#pragma unroll 8
    for (int kt = 0; kt < N_CELL / 32; ++kt) {
      const int k0 = kt * 32;
      V16BF A, B;
      A.q[0] = *(const u32x4*)(Abase + k0);
      A.q[1] = *(const u32x4*)(Abase + k0 + 16);
      B.q[0] = *(const u32x4*)(Bbase + k0);
      B.q[1] = *(const u32x4*)(Bbase + k0 + 16);
      // WGP-scope prefetch (default scope pulls into all cache levels, unlike
      // SCOPE_SE/SYS which stop at GL2): stream B ~16 iterations ahead into L0.
      asm volatile("global_prefetch_b8 %0, off" :: "v"(Bbase + k0 + 512));
      acc = __builtin_amdgcn_wmma_f32_16x16x32_bf16(
                false, A.v, false, B.v, (short)0, acc, false, false);
    }

    const int phase = t / T_EPOCH;   // 0 fix,1 sample,2 delay,3 test,4 response
    if (lane < 16) {
      // noise is single-use streaming data: keep it out of the resident L2 set
      const float nz = 0.1f * __builtin_nontemporal_load(noise + (size_t)t * N_CELL + nn);
#pragma unroll
      for (int m = 0; m < 4; ++m) {
        const float e = (phase == 1) ? dsv[m] : ((phase == 3) ? dtv[m] : 0.f);
        float p = acc[m] + bias + e;            // C-layout: VGPR m, lanes 0..15 = M=m
        p = fmaxf(p, 0.f);
        rr[m] += (-rr[m] + p + nz) * 0.2f;      // dt/tau = 0.2
        Ab[m * N_CELL + nn] = f2bf(rr[m]);      // publish next-step state (bf16)
      }
      if (phase == 2) {
#pragma unroll
        for (int m = 0; m < 4; ++m) { rmv[2*m] += rr[m]*mw0; rmv[2*m+1] += rr[m]*mw1; }
      } else if (phase == 4) {
#pragma unroll
        for (int m = 0; m < 4; ++m) { rsv[2*m] += rr[m]*ow0; rsv[2*m+1] += rr[m]*ow1; }
      }
    }

    // ---- grid-wide step barrier (monotonic counter; 256 single-wave blocks,
    //      all trivially co-resident; __syncthreads is a NOP for 1-wave groups)
    __threadfence();                 // release my state writes to device scope
    __syncthreads();
    if (threadIdx.x == 0) {
      atomicAdd(counter, 1u);
      const unsigned tgt = (unsigned)(t + 1) * gridDim.x;
      while (atomicAdd(counter, 0u) < tgt) { __builtin_amdgcn_s_sleep(2); }
    }
    __syncthreads();
    __threadfence();                 // acquire: invalidate L0 before reading peers' state
  }

  // ---- final readout reduction across the 16 live lanes (others hold zeros)
  float vals[16];
#pragma unroll
  for (int i = 0; i < 8; ++i) { vals[i] = rmv[i]; vals[8 + i] = rsv[i]; }
#pragma unroll
  for (int i = 0; i < 16; ++i) {
    float v = vals[i];
    v += __shfl_xor(v, 16, 32);
    v += __shfl_xor(v,  8, 32);
    v += __shfl_xor(v,  4, 32);
    v += __shfl_xor(v,  2, 32);
    v += __shfl_xor(v,  1, 32);
    if (lane == 0) atomicAdd(&accum[i], v);
  }
}

// ---------------------------------------------------------------- finalize
__global__ void rnn_final(const float* __restrict__ accum, float* __restrict__ out) {
  const int i = threadIdx.x;
  if (i < 16) out[i] = accum[i] * (1.0f / (float)T_EPOCH);   // epoch mean
}

// ---------------------------------------------------------------- launch
extern "C" void kernel_launch(void* const* d_in, const int* in_sizes, int n_in,
                              void* d_out, int out_size, void* d_ws, size_t ws_size,
                              hipStream_t stream) {
  const float* x     = (const float*)d_in[0];   // [4,2,2]
  const float* rec_w = (const float*)d_in[1];   // [N,N]
  const float* rec_b = (const float*)d_in[2];   // [N]
  const float* inp_w = (const float*)d_in[3];   // [N,2]
  const float* out_w = (const float*)d_in[4];   // [2,N]
  const float* mem_w = (const float*)d_in[5];   // [2,N]
  const float* noise = (const float*)d_in[6];   // [2500,N]

  char* ws = (char*)d_ws;
  unsigned*       counter = (unsigned*)(ws + 0);
  float*          accum   = (float*)(ws + 256);
  float*          drv_s   = (float*)(ws + 1024);                 // 4*N f32
  float*          drv_t   = drv_s + 4 * N_CELL;                  // 4*N f32
  unsigned short* Ab      = (unsigned short*)(ws + 132096);      // 4*N bf16 (32KB)
  unsigned short* Wb      = (unsigned short*)(ws + 263168);      // N*N bf16
  const size_t needed = 263168 + (size_t)N_CELL * N_CELL * 2;
  if (ws_size < needed) return;

  rnn_init <<<1, 256, 0, stream>>>(counter, accum, (unsigned*)Ab);
  rnn_drive<<<N_CELL / 256, 256, 0, stream>>>(x, inp_w, drv_s, drv_t);
  rnn_cvt  <<<(N_CELL * N_CELL) / (256 * 8), 256, 0, stream>>>(rec_w, Wb);
  rnn_persist<<<NBLOCKS, NTHREADS, 0, stream>>>(Wb, Ab, rec_b, drv_s, drv_t,
                                                mem_w, out_w, noise, counter, accum);
  rnn_final<<<1, 16, 0, stream>>>(accum, (float*)d_out);
}
